// ModulatedConv2d_1108101563041
// MI455X (gfx1250) — matrix operand order repaired and checked by reading, not compiled, not run on hardware
//
#include <hip/hip_runtime.h>
#include <hip/hip_bf16.h>

// Shapes from the reference
#define BATCH    8
#define CIN      512
#define COUT     512
#define SDIM     512
#define IMH      64
#define IMW      64
#define NPIX     (IMH * IMW)

#define MOD_SCALE   0.04419417382415922f   // 1/sqrt(512)
#define CONV_SCALE  0.014731391274719742f  // 1/sqrt(512*9)
#define DEMOD_EPS   1e-8f

// Halo tile geometry: 2 output rows -> 4 input rows x 66 cols
#define HALO_W   66
#define HALO_H   4
#define BPOS     (HALO_H * HALO_W)   // 264 positions

typedef __attribute__((ext_vector_type(16))) __bf16 v16bf;
typedef __attribute__((ext_vector_type(8)))  float  v8f;
typedef int v4i_g __attribute__((vector_size(16)));   // gcc-style int4 (builtin param type)

#define AS1 __attribute__((address_space(1)))
#define AS3 __attribute__((address_space(3)))

#if __has_builtin(__builtin_amdgcn_global_load_async_to_lds_b128) && \
    __has_builtin(__builtin_amdgcn_s_wait_asynccnt)
#define USE_ASYNC_LDS 1
#else
#define USE_ASYNC_LDS 0
#endif

union FragU {
    v16bf v;
    uint4 q[2];
};

// ---------------------------------------------------------------------------
// Kernel 1: s[b,c] = style[b,:] @ (mod_weight * MOD_SCALE).T + mod_bias[c]
// ---------------------------------------------------------------------------
__global__ __launch_bounds__(256) void style_mod_kernel(
    const float* __restrict__ style,      // [B, SDIM]
    const float* __restrict__ mod_weight, // [CIN, SDIM]
    const float* __restrict__ mod_bias,   // [CIN]
    float* __restrict__ s_out)            // [B, CIN]
{
    int tid = blockIdx.x * blockDim.x + threadIdx.x;
    if (tid >= BATCH * CIN) return;
    int b = tid >> 9;
    int c = tid & (CIN - 1);
    const float* st = style + b * SDIM;
    const float* mw = mod_weight + c * SDIM;
    float acc = 0.0f;
    #pragma unroll 8
    for (int j = 0; j < SDIM; ++j) acc += st[j] * mw[j];
    s_out[tid] = acc * MOD_SCALE + mod_bias[c];
}

// ---------------------------------------------------------------------------
// Kernel 2: demod[b,o] = rsqrt(CONV_SCALE^2 * sum_c s^2 * sum_k w^2 + eps)
// ---------------------------------------------------------------------------
__global__ __launch_bounds__(256) void demod_kernel(
    const float* __restrict__ weight,  // [COUT, CIN, 9]
    const float* __restrict__ s_in,    // [B, CIN]
    float* __restrict__ demod_out)     // [B, COUT]
{
    int tid = blockIdx.x * blockDim.x + threadIdx.x;
    if (tid >= BATCH * COUT) return;
    int b = tid >> 9;
    int o = tid & (COUT - 1);
    const float* sb = s_in + b * CIN;
    const float* wo = weight + (size_t)o * CIN * 9;
    float acc = 0.0f;
    for (int c = 0; c < CIN; ++c) {
        float sv = sb[c];
        float ss = sv * sv;
        float wsum = 0.0f;
        #pragma unroll
        for (int k = 0; k < 9; ++k) {
            float w = wo[c * 9 + k];
            wsum += w * w;
        }
        acc += ss * wsum;
    }
    demod_out[tid] = rsqrtf(CONV_SCALE * CONV_SCALE * acc + DEMOD_EPS);
}

// ---------------------------------------------------------------------------
// Kernel 2b: transpose raw conv weight to bf16, tap-major:
//   wbf[kk][o*512+c] = (bf16) weight[(o*512+c)*9 + kk]
// ---------------------------------------------------------------------------
__global__ __launch_bounds__(256) void wcvt_kernel(
    const float* __restrict__ weight,  // [COUT*CIN, 9]
    __bf16* __restrict__ wbf)          // [9, COUT*CIN]
{
    int e = blockIdx.x * 256 + threadIdx.x;   // 0 .. 9*512*512-1
    int kk = e >> 18;                         // plane (2^18 = 512*512)
    int oc = e & (COUT * CIN - 1);
    wbf[e] = (__bf16)weight[(size_t)oc * 9 + kk];
}

// ---------------------------------------------------------------------------
// Kernel 3: implicit-GEMM modulated conv with v_wmma_f32_16x16x32_bf16.
// Block: 256 threads (8 waves). Tile: 64 out-ch x 128 pixels (2 image rows).
// 16 chunks of 32 input channels; per chunk:
//   - A: 9-tap bf16 weight tile -> LDS via global_load_async_to_lds_b128
//        (ASYNCcnt-tracked DMA; falls back to b128 copy if builtin absent)
//   - B: interior halo (4 rows x 64 cols) staged float4->bf16, scaled by
//        conv_scale*s[b,c]; edge cols / OOB rows stay pre-zeroed
//   - 9 taps x (2x2) WMMA per wave = 36 WMMAs per barrier pair.
// demod[b,o] folded into the output store.
// ---------------------------------------------------------------------------
__global__ __launch_bounds__(256) void modconv_wmma_kernel(
    const float*  __restrict__ input,  // [B, CIN, 64, 64]
    const __bf16* __restrict__ wbf,    // [9, COUT, CIN]
    const float*  __restrict__ s_in,   // [B, CIN]
    const float*  __restrict__ demod,  // [B, COUT]
    float* __restrict__ out)           // [B, COUT, 64, 64]
{
    __shared__ __align__(16) __bf16 Alds[9 * 64 * 32];   // 36864 B
    __shared__ __align__(16) __bf16 Blds[BPOS * 32];     // 16896 B

    const int t    = threadIdx.x;
    const int lane = t & 31;
    const int wv   = t >> 5;
    const int b    = blockIdx.z;
    const int o0   = blockIdx.y * 64;
    const int h0   = blockIdx.x * 2;          // two image rows per tile
    const int p0   = h0 * IMW;                // 128 pixels

    const int mo   = (wv & 1) * 32;           // wave o-offset in tile
    const int no   = (wv >> 1) * 32;          // wave pixel-offset in tile
    const int lrow = lane & 15;
    const int hi16 = lane >> 4;
    const int g0   = hi16 * 8;                // A fragment element offset
    const int bb   = hi16 * 16;               // unused; B fragments also use g0

    // ---- pre-zero Blds once: edge columns & OOB rows stay zero forever ----
    {
        uint4 z = {0u, 0u, 0u, 0u};
        uint4* bz = (uint4*)Blds;
        #pragma unroll
        for (int i = 0; i < 9; ++i) {
            int q = i * 256 + t;
            if (q < (BPOS * 32) / 8) bz[q] = z;   // 1056 uint4
        }
    }

    // ---- B-stage per-thread constants (all shifts) ----
    const int hl    = (t & 63) >> 4;          // halo row 0..3
    const int w4    = t & 15;                 // float4 column
    const int cquad = t >> 6;                 // 0..3
    const int hh    = h0 - 1 + hl;            // image row (may be OOB)
    const bool hok  = (hh >= 0) && (hh < IMH);
    const int bldsb = (hl * HALO_W + w4 * 4 + 1) * 32;  // LDS pos base

    // Per-lane halo position base for each B sub-tile (center tap)
    int nbase[2];
    #pragma unroll
    for (int ni = 0; ni < 2; ++ni) {
        int n = no + ni * 16 + lrow;          // tile pixel 0..127
        nbase[ni] = (1 + (n >> 6)) * HALO_W + 1 + (n & 63);
    }

    const float* sb = s_in + b * CIN;
    const float* xrow = input + (size_t)b * CIN * NPIX + (size_t)hh * IMW + w4 * 4;

    v8f acc[2][2] = {};

    for (int c0 = 0; c0 < CIN; c0 += 32) {
        __syncthreads();   // protect LDS from previous stage's readers

        // ---- stage A: 9 x b128 weight-tile copy (async DMA if available) --
        {
            const __bf16* wb0 = wbf + (size_t)o0 * CIN + c0;
            uint4* adst = (uint4*)Alds;
            #pragma unroll
            for (int i = 0; i < 9; ++i) {
                int q   = i * 256 + t;        // 0..2303
                int row = q >> 2;             // 0..575  (= kk*64 + o)
                int col = q & 3;
                int kk  = row >> 6;
                int o   = row & 63;
                const uint4* src =
                    (const uint4*)(wb0 + (size_t)kk * (COUT * CIN) + (size_t)o * CIN) + col;
#if USE_ASYNC_LDS
                __builtin_amdgcn_global_load_async_to_lds_b128(
                    (AS1 v4i_g*)src, (AS3 v4i_g*)&adst[q], 0, 0);
#else
                adst[q] = *src;
#endif
            }
        }

        // ---- stage B: interior halo, float4 loads, scaled converts ----
        if (hok) {
            #pragma unroll
            for (int i = 0; i < 8; ++i) {
                int cc = i * 4 + cquad;       // 0..31
                const float4 v4 =
                    *(const float4*)(xrow + (size_t)(c0 + cc) * NPIX);
                float sc = CONV_SCALE * sb[c0 + cc];
                Blds[bldsb + cc + 0 * 32] = (__bf16)(v4.x * sc);
                Blds[bldsb + cc + 1 * 32] = (__bf16)(v4.y * sc);
                Blds[bldsb + cc + 2 * 32] = (__bf16)(v4.z * sc);
                Blds[bldsb + cc + 3 * 32] = (__bf16)(v4.w * sc);
            }
        }

#if USE_ASYNC_LDS
        __builtin_amdgcn_s_wait_asynccnt(0);   // own async DMAs done in LDS
#endif
        __syncthreads();                        // publish LDS to all waves

        // ---- compute: 9 taps x 2x2 WMMA ----
        #pragma unroll
        for (int kk = 0; kk < 9; ++kk) {
            const int dh = kk / 3 - 1;
            const int dw = kk % 3 - 1;
            const int shift = dh * HALO_W + dw;

            FragU afr[2], bfr[2];
            #pragma unroll
            for (int mi = 0; mi < 2; ++mi) {
                const __bf16* ap = &Alds[(kk * 64 + mo + mi * 16 + lrow) * 32];
                afr[mi].q[0] = *(const uint4*)(ap + g0);       // K = g0..g0+7
                afr[mi].q[1] = *(const uint4*)(ap + g0 + 16);  // K = 16+g0..
            }
            #pragma unroll
            for (int ni = 0; ni < 2; ++ni) {
                const __bf16* bp = &Blds[(nbase[ni] + shift) * 32];
                bfr[ni].q[0] = *(const uint4*)(bp + g0);       // K = g0..g0+7
                bfr[ni].q[1] = *(const uint4*)(bp + g0 + 16);  // K = 16+g0..
            }

            #pragma unroll
            for (int mi = 0; mi < 2; ++mi)
                #pragma unroll
                for (int ni = 0; ni < 2; ++ni)
                    acc[mi][ni] = __builtin_amdgcn_wmma_f32_16x16x32_bf16(
                        false, afr[mi].v, false, bfr[ni].v,
                        (short)0, acc[mi][ni], false, false);
        }
    }

    // ---- store, folding demod[b,o]; N across lanes -> coalesced ----
    const float* db = demod + b * COUT;
    float* ob = out + (size_t)b * COUT * NPIX;
    #pragma unroll
    for (int mi = 0; mi < 2; ++mi) {
        #pragma unroll
        for (int r = 0; r < 8; ++r) {
            int o = o0 + mo + mi * 16 + hi16 * 8 + r;
            float dm = db[o];
            #pragma unroll
            for (int ni = 0; ni < 2; ++ni) {
                int p = p0 + no + ni * 16 + (lane & 15);
                ob[(size_t)o * NPIX + p] = acc[mi][ni][r] * dm;
            }
        }
    }
}

// ---------------------------------------------------------------------------
extern "C" void kernel_launch(void* const* d_in, const int* in_sizes, int n_in,
                              void* d_out, int out_size, void* d_ws, size_t ws_size,
                              hipStream_t stream) {
    const float* input      = (const float*)d_in[0]; // [8,512,64,64]
    const float* style      = (const float*)d_in[1]; // [8,512]
    const float* weight     = (const float*)d_in[2]; // [1,512,512,3,3]
    const float* mod_weight = (const float*)d_in[3]; // [512,512]
    const float* mod_bias   = (const float*)d_in[4]; // [512]
    float* out = (float*)d_out;

    // workspace layout: s[8*512] f32 | demod[8*512] f32 | wbf[9*512*512] bf16
    float*  s_ws     = (float*)d_ws;
    float*  demod_ws = s_ws + BATCH * CIN;
    __bf16* wbf_ws   = (__bf16*)(s_ws + 2 * BATCH * CIN);

    style_mod_kernel<<<(BATCH * CIN + 255) / 256, 256, 0, stream>>>(
        style, mod_weight, mod_bias, s_ws);

    demod_kernel<<<(BATCH * COUT + 255) / 256, 256, 0, stream>>>(
        weight, s_ws, demod_ws);

    wcvt_kernel<<<(9 * COUT * CIN) / 256, 256, 0, stream>>>(
        weight, wbf_ws);

    dim3 grid(IMH / 2, COUT / 64, BATCH);  // 32 x 8 x 8 = 2048 blocks
    modconv_wmma_kernel<<<grid, 256, 0, stream>>>(
        input, wbf_ws, s_ws, demod_ws, out);
}
